// GAU_11982958756278
// MI455X (gfx1250) — matrix operand order repaired and checked by reading, not compiled, not run on hardware
//
#include <hip/hip_runtime.h>

// ---------------------------------------------------------------------------
// GAU block for MI455X (gfx1250, wave32, WMMA bf16 16x16x32, f32 accumulate)
// All WMMA operands laid out so each lane fragment is contiguous 16B loads:
//   A row-major [M][K], B stored K-contiguous (weights transposed, v transposed)
// ---------------------------------------------------------------------------

#define BATCH   4
#define SEQ     2048
#define DIM     512
#define QK      128
#define HID     1024
#define HID2    2048
#define ROWS    (BATCH * SEQ)            // 8192
#define SCALE_B 11.313708498984760f      // sqrt(128)
#define INV_LOG16_X8 2.8853900817779268f // 8 / ln(16)
#define LN10000 9.210340371976184f       // ln(10000)

typedef __attribute__((ext_vector_type(16))) __bf16 v16bf;
typedef __attribute__((ext_vector_type(8)))  __bf16 v8bf;
typedef __attribute__((ext_vector_type(8)))  float  v8f;

static __device__ __forceinline__ __bf16 to_bf16(float f) {
    unsigned u = __builtin_bit_cast(unsigned, f);
    u += 0x7FFFu + ((u >> 16) & 1u);          // round-to-nearest-even
    unsigned short h = (unsigned short)(u >> 16);
    return __builtin_bit_cast(__bf16, h);
}

static __device__ __forceinline__ float fast_silu(float x) {
    // x * sigmoid(x), v_rcp_f32 instead of IEEE divide
    return x * __builtin_amdgcn_rcpf(1.0f + __expf(-x));
}

static __device__ __forceinline__ v8f wmma_bf16(v16bf a, v16bf b, v8f c) {
    return __builtin_amdgcn_wmma_f32_16x16x32_bf16(
        false, a, false, b, (short)0, c, false, false);
}

static __device__ __forceinline__ v16bf cat8(v8bf lo, v8bf hi) {
    return __builtin_shufflevector(lo, hi, 0, 1, 2, 3, 4, 5, 6, 7,
                                            8, 9, 10, 11, 12, 13, 14, 15);
}

// A-fragment (16x32 bf16): lane m=lane&15 is row, half=lane>>4.
// elements 0..7 -> K = kb + half*8 + e ; elements 8..15 -> K = kb+16+half*8+e-8
// => two aligned contiguous 16-byte loads per lane.
static __device__ __forceinline__ v16bf load_a_frag(const __bf16* __restrict__ base,
                                                    int ld, int mrow, int kb, int lane) {
    const int half = lane >> 4, m = lane & 15;
    const __bf16* p = base + (size_t)(mrow + m) * ld + kb + half * 8;
    return cat8(*(const v8bf*)p, *(const v8bf*)(p + 16));
}

// B-fragment (32x16 bf16) from K-CONTIGUOUS storage (row n holds the K run):
// lane n=lane&15 is column; elements 0..7 -> K = kb + half*8 + e ; 8..15 -> K = kb+16+half*8+e-8
// => two aligned 16-byte loads per lane.
static __device__ __forceinline__ v16bf load_bk_frag(const __bf16* __restrict__ base,
                                                     int ld, int col, int kb, int lane) {
    const int half = lane >> 4, n = lane & 15;
    const __bf16* p = base + (size_t)(col + n) * ld + kb + half * 8;
    return cat8(*(const v8bf*)p, *(const v8bf*)(p + 16));
}

// ---------------------------------------------------------------------------
// 1) LayerNorm: one row (512 f32) per 256-thread block
// ---------------------------------------------------------------------------
__global__ void ln_kernel(const float* __restrict__ x, const float* __restrict__ g,
                          const float* __restrict__ b, float* __restrict__ nx) {
    const int row = blockIdx.x;
    const int tid = threadIdx.x;
    const float* xr = x + (size_t)row * DIM;
    float v0 = xr[tid], v1 = xr[tid + 256];
    __shared__ float rs[256], rss[256];
    rs[tid]  = v0 + v1;
    rss[tid] = v0 * v0 + v1 * v1;
    __syncthreads();
#pragma unroll
    for (int off = 128; off > 0; off >>= 1) {
        if (tid < off) { rs[tid] += rs[tid + off]; rss[tid] += rss[tid + off]; }
        __syncthreads();
    }
    const float mu   = rs[0] * (1.0f / DIM);
    const float var  = rss[0] * (1.0f / DIM) - mu * mu;
    const float rstd = rsqrtf(var + 1e-5f);
    float* o = nx + (size_t)row * DIM;
    o[tid]       = (v0 - mu) * rstd * g[tid]       + b[tid];
    o[tid + 256] = (v1 - mu) * rstd * g[tid + 256] + b[tid + 256];
}

// ---------------------------------------------------------------------------
// 2) Token-shift (first 256 channels from previous position) + bf16 convert
// ---------------------------------------------------------------------------
__global__ void shift_cvt_kernel(const float* __restrict__ nx, __bf16* __restrict__ nxb) {
    const int idx = blockIdx.x * blockDim.x + threadIdx.x;   // ROWS*DIM threads
    const int row = idx >> 9, c = idx & 511;
    const int n = row & (SEQ - 1);
    float v;
    if (c < DIM / 2) v = (n == 0) ? 0.0f : nx[(size_t)(row - 1) * DIM + c];
    else             v = nx[(size_t)row * DIM + c];
    nxb[idx] = to_bf16(v);
}

// Convert + transpose weight: w[K][N] (f32) -> wt[N][K] (bf16), output-linear
__global__ void cvt_transpose_kernel(const float* __restrict__ w, __bf16* __restrict__ wt,
                                     int K, int N) {
    const int idx = blockIdx.x * blockDim.x + threadIdx.x;
    if (idx >= K * N) return;
    const int n = idx / K, k = idx - n * K;
    wt[idx] = to_bf16(w[(size_t)k * N + n]);
}

// ---------------------------------------------------------------------------
// 3) GEMM hid: [8192x512] x [512x2048], silu, split into v^T(bf16)/gate(f32)
//    Wave tile: 32(M) x 64(N); A reused across 4 B frags, B reused across 2 A.
// ---------------------------------------------------------------------------
__global__ void gemm_hid_kernel(const __bf16* __restrict__ nxb, const __bf16* __restrict__ wt,
                                const float* __restrict__ bias,
                                __bf16* __restrict__ vT, float* __restrict__ gate) {
    const int lane = threadIdx.x & 31, wave = threadIdx.x >> 5;
    const int wid  = blockIdx.x * 8 + wave;
    const int mt = wid / (HID2 / 64), nt = wid % (HID2 / 64);
    const int mrow = mt * 32, ncol = nt * 64;
    v8f c[2][4];
#pragma unroll
    for (int m = 0; m < 2; ++m)
#pragma unroll
        for (int t = 0; t < 4; ++t)
#pragma unroll
            for (int i = 0; i < 8; ++i) c[m][t][i] = 0.0f;

    for (int kb = 0; kb < DIM; kb += 32) {
        const v16bf a0 = load_a_frag(nxb, DIM, mrow,      kb, lane);
        const v16bf a1 = load_a_frag(nxb, DIM, mrow + 16, kb, lane);
#pragma unroll
        for (int t = 0; t < 4; ++t) {
            const v16bf b = load_bk_frag(wt, DIM, ncol + t * 16, kb, lane);
            c[0][t] = wmma_bf16(a0, b, c[0][t]);
            c[1][t] = wmma_bf16(a1, b, c[1][t]);
        }
    }
    const int half = lane >> 4, n16 = lane & 15;
#pragma unroll
    for (int t = 0; t < 4; ++t) {
        const int j = ncol + t * 16 + n16;
        const float bj = bias[j];
        const bool is_v = (ncol + t * 16) < HID;    // wave-uniform branch
#pragma unroll
        for (int m = 0; m < 2; ++m) {
            if (is_v) {
                // transposed store: 8 consecutive tokens per lane (16B run)
                __bf16* vp = vT + (size_t)j * ROWS + mrow + m * 16 + 8 * half;
#pragma unroll
                for (int r = 0; r < 8; ++r) vp[r] = to_bf16(fast_silu(c[m][t][r] + bj));
            } else {
#pragma unroll
                for (int r = 0; r < 8; ++r) {
                    const int row = mrow + m * 16 + r + 8 * half;
                    gate[(size_t)row * HID + (j - HID)] = fast_silu(c[m][t][r] + bj);
                }
            }
        }
    }
}

// ---------------------------------------------------------------------------
// 4) GEMM qk: [8192x512] x [512x128], silu, gamma/beta -> q,k, fused RoPE.
//    RoPE pair partner lives in lane^1 of the WMMA C layout -> one shfl_xor.
// ---------------------------------------------------------------------------
__global__ void gemm_qk_kernel(const __bf16* __restrict__ nxb, const __bf16* __restrict__ wt,
                               const float* __restrict__ bq,
                               const float* __restrict__ osg, const float* __restrict__ osb,
                               __bf16* __restrict__ qbuf, __bf16* __restrict__ kbuf) {
    const int lane = threadIdx.x & 31, wave = threadIdx.x >> 5;
    const int wid  = blockIdx.x * 8 + wave;
    const int mt = wid / 2, nt = wid % 2;
    const int mrow = mt * 32, ncol = nt * 64;
    v8f c[2][4];
#pragma unroll
    for (int m = 0; m < 2; ++m)
#pragma unroll
        for (int t = 0; t < 4; ++t)
#pragma unroll
            for (int i = 0; i < 8; ++i) c[m][t][i] = 0.0f;

    for (int kb = 0; kb < DIM; kb += 32) {
        const v16bf a0 = load_a_frag(nxb, DIM, mrow,      kb, lane);
        const v16bf a1 = load_a_frag(nxb, DIM, mrow + 16, kb, lane);
#pragma unroll
        for (int t = 0; t < 4; ++t) {
            const v16bf b = load_bk_frag(wt, DIM, ncol + t * 16, kb, lane);
            c[0][t] = wmma_bf16(a0, b, c[0][t]);
            c[1][t] = wmma_bf16(a1, b, c[1][t]);
        }
    }
    const int half = lane >> 4, n16 = lane & 15;
#pragma unroll
    for (int t = 0; t < 4; ++t) {
        const int j = ncol + t * 16 + n16;
        const float bj = bq[j];
        const float g0 = osg[j],      b0 = osb[j];
        const float g1 = osg[QK + j], b1 = osb[QK + j];
        const bool  rot = (ncol + t * 16) < 32;             // wave-uniform
        const float invf = __expf(-(float)(j & ~1) * (LN10000 / 32.0f));
        const float sgn  = (j & 1) ? 1.0f : -1.0f;
#pragma unroll
        for (int m = 0; m < 2; ++m)
#pragma unroll
        for (int r = 0; r < 8; ++r) {
            const int row = mrow + m * 16 + r + 8 * half;
            const float s = fast_silu(c[m][t][r] + bj);
            float qv = s * g0 + b0;
            float kv = s * g1 + b1;
            const float qp = __shfl_xor(qv, 1, 32);         // RoPE pair partner
            const float kp = __shfl_xor(kv, 1, 32);
            if (rot) {
                const float fr = (float)(row & (SEQ - 1)) * invf;
                const float cf = __cosf(fr), sf = __sinf(fr);
                qv = qv * cf + sgn * qp * sf;
                kv = kv * cf + sgn * kp * sf;
            }
            qbuf[(size_t)row * QK + j] = to_bf16(qv);
            kbuf[(size_t)row * QK + j] = to_bf16(kv);
        }
    }
}

// ---------------------------------------------------------------------------
// 5) Fused attention: block = (batch, 16 query rows), 8 waves.
//    chunk loop over 128 keys: wave w computes sim tile for keys w*16..+15
//    (bias + relu^2/N + mask fused), parks bf16 attn in LDS; then every wave
//    runs attn(16x128) @ v^T(128 x its 128-wide HID slice). Gate fused out.
//    The NxN attention matrix never touches HBM.
// ---------------------------------------------------------------------------
__global__ void attn_kernel(const __bf16* __restrict__ qbuf, const __bf16* __restrict__ kbuf,
                            const __bf16* __restrict__ vT, const float* __restrict__ gate,
                            const float* __restrict__ rel_emb,
                            const unsigned char* __restrict__ mask,
                            __bf16* __restrict__ omb) {
    const int lane = threadIdx.x & 31, wave = threadIdx.x >> 5;
    const int half = lane >> 4, n16 = lane & 15;
    const int bb = blockIdx.x >> 7;          // batch
    const int q0 = (blockIdx.x & 127) * 16;  // seq-local query base
    const int grow = bb * SEQ + q0;          // global row base

    __shared__ __bf16 attn_s[16][128];

    v16bf aq[4];                              // q fragments, K=128 -> 4 frags
#pragma unroll
    for (int f = 0; f < 4; ++f) aq[f] = load_a_frag(qbuf, QK, grow, f * 32, lane);

    v8f acc[8];
#pragma unroll
    for (int nt = 0; nt < 8; ++nt)
#pragma unroll
        for (int i = 0; i < 8; ++i) acc[nt][i] = 0.0f;

    const int hb = wave * 128;                // this wave's HID slice
    const __bf16* kmat = kbuf + (size_t)bb * SEQ * QK;

    for (int chunk = 0; chunk < SEQ / 128; ++chunk) {
        // ---- phase 1: sim tile (16 queries x 16 keys) for this wave ----
        const int jb = chunk * 128 + wave * 16;
        v8f s;
#pragma unroll
        for (int i = 0; i < 8; ++i) s[i] = 0.0f;
#pragma unroll
        for (int f = 0; f < 4; ++f) {
            const v16bf bk = load_bk_frag(kmat, QK, jb, f * 32, lane);  // k^T
            s = wmma_bf16(aq[f], bk, s);
        }
        const int kj = jb + n16;
        const float mk = mask[(size_t)bb * SEQ + kj] ? 1.0f : 0.0f;
#pragma unroll
        for (int r = 0; r < 8; ++r) {
            const int qi = q0 + r + 8 * half;
            const int nrel = qi - kj;                     // n = -(j-i)
            const int ret = (nrel < 0) ? 16 : 0;
            const int na = nrel < 0 ? -nrel : nrel;
            int idx;
            if (na < 8) idx = ret + na;
            else {
                int lv = 8 + (int)(__logf((float)na * 0.125f) * INV_LOG16_X8);
                idx = ret + (lv > 15 ? 15 : lv);
            }
            const float bias = rel_emb[idx] * SCALE_B;
            float t = (s[r] + bias) * (1.0f / (float)SEQ);
            t = t > 0.0f ? t : 0.0f;
            attn_s[r + 8 * half][wave * 16 + n16] = to_bf16(t * t * mk);
        }
        __syncthreads();
        // ---- phase 2: acc += attn_chunk(16x128) @ v_chunk(128 x 128) ----
        v16bf aa[4];
#pragma unroll
        for (int f = 0; f < 4; ++f) {                      // ds_load_b128 x2
            const __bf16* p = &attn_s[n16][f * 32 + half * 8];
            aa[f] = cat8(*(const v8bf*)p, *(const v8bf*)(p + 16));
        }
        // v^T[hid][token]: per lane hid = hb + nt*16 + n16, token run contiguous
        const int tok0 = bb * SEQ + chunk * 128;
#pragma unroll
        for (int nt = 0; nt < 8; ++nt) {
#pragma unroll
            for (int f = 0; f < 4; ++f) {
                const __bf16* p = vT + (size_t)(hb + nt * 16 + n16) * ROWS
                                     + tok0 + f * 32 + half * 8;
                const v16bf bv = cat8(*(const v8bf*)p, *(const v8bf*)(p + 16));
                acc[nt] = wmma_bf16(aa[f], bv, acc[nt]);
            }
        }
        __syncthreads();
    }
    // ---- epilogue: gate and store bf16 (feeds output GEMM) ----
#pragma unroll
    for (int nt = 0; nt < 8; ++nt) {
        const int col = hb + nt * 16 + n16;
#pragma unroll
        for (int r = 0; r < 8; ++r) {
            const size_t row = (size_t)grow + r + 8 * half;
            const float g = gate[row * HID + col];
            omb[row * HID + col] = to_bf16(acc[nt][r] * g);
        }
    }
}

// ---------------------------------------------------------------------------
// 6) Output GEMM: [8192x1024] x [1024x512] + b_out + residual x
// ---------------------------------------------------------------------------
__global__ void gemm_out_kernel(const __bf16* __restrict__ omb, const __bf16* __restrict__ wt,
                                const float* __restrict__ bias, const float* __restrict__ x,
                                float* __restrict__ out) {
    const int lane = threadIdx.x & 31, wave = threadIdx.x >> 5;
    const int wid  = blockIdx.x * 8 + wave;
    const int mt = wid / (DIM / 64), nt = wid % (DIM / 64);
    const int mrow = mt * 32, ncol = nt * 64;
    v8f c[2][4];
#pragma unroll
    for (int m = 0; m < 2; ++m)
#pragma unroll
        for (int t = 0; t < 4; ++t)
#pragma unroll
            for (int i = 0; i < 8; ++i) c[m][t][i] = 0.0f;

    for (int kb = 0; kb < HID; kb += 32) {
        const v16bf a0 = load_a_frag(omb, HID, mrow,      kb, lane);
        const v16bf a1 = load_a_frag(omb, HID, mrow + 16, kb, lane);
#pragma unroll
        for (int t = 0; t < 4; ++t) {
            const v16bf b = load_bk_frag(wt, HID, ncol + t * 16, kb, lane);
            c[0][t] = wmma_bf16(a0, b, c[0][t]);
            c[1][t] = wmma_bf16(a1, b, c[1][t]);
        }
    }
    const int half = lane >> 4, n16 = lane & 15;
#pragma unroll
    for (int t = 0; t < 4; ++t) {
        const int j = ncol + t * 16 + n16;
        const float bj = bias[j];
#pragma unroll
        for (int m = 0; m < 2; ++m)
#pragma unroll
        for (int r = 0; r < 8; ++r) {
            const size_t row = (size_t)mrow + m * 16 + r + 8 * half;
            out[row * DIM + j] = c[m][t][r] + bj + x[row * DIM + j];
        }
    }
}

// ---------------------------------------------------------------------------
// Host launcher
// ---------------------------------------------------------------------------
extern "C" void kernel_launch(void* const* d_in, const int* in_sizes, int n_in,
                              void* d_out, int out_size, void* d_ws, size_t ws_size,
                              hipStream_t stream) {
    const float* x      = (const float*)d_in[0];
    const float* ln_g   = (const float*)d_in[1];
    const float* ln_b   = (const float*)d_in[2];
    const float* w_hid  = (const float*)d_in[3];
    const float* b_hid  = (const float*)d_in[4];
    const float* w_qk   = (const float*)d_in[5];
    const float* b_qk   = (const float*)d_in[6];
    const float* osg    = (const float*)d_in[7];
    const float* osb    = (const float*)d_in[8];
    const float* w_out  = (const float*)d_in[9];
    const float* b_out  = (const float*)d_in[10];
    const float* rel    = (const float*)d_in[11];
    const unsigned char* mask = (const unsigned char*)d_in[12];
    float* out = (float*)d_out;

    char* ws = (char*)d_ws;
    size_t off = 0;
    auto carve = [&](size_t bytes) { char* p = ws + off; off = (off + bytes + 255) & ~(size_t)255; return p; };

    float*  nx    = (float*) carve((size_t)ROWS * DIM * 4);
    __bf16* nxb   = (__bf16*)carve((size_t)ROWS * DIM * 2);
    __bf16* whT   = (__bf16*)carve((size_t)DIM * HID2 * 2);   // [2048][512]
    __bf16* wqT   = (__bf16*)carve((size_t)DIM * QK * 2);     // [128][512]
    __bf16* woT   = (__bf16*)carve((size_t)HID * DIM * 2);    // [512][1024]
    __bf16* vT    = (__bf16*)carve((size_t)ROWS * HID * 2);   // [1024][8192]
    float*  gate  = (float*) carve((size_t)ROWS * HID * 4);
    __bf16* qbuf  = (__bf16*)carve((size_t)ROWS * QK * 2);
    __bf16* kbuf  = (__bf16*)carve((size_t)ROWS * QK * 2);
    __bf16* omb   = (__bf16*)carve((size_t)ROWS * HID * 2);
    (void)ws_size; (void)n_in; (void)in_sizes; (void)out_size;

    ln_kernel<<<ROWS, 256, 0, stream>>>(x, ln_g, ln_b, nx);
    shift_cvt_kernel<<<(ROWS * DIM) / 256, 256, 0, stream>>>(nx, nxb);
    cvt_transpose_kernel<<<(DIM * HID2 + 255) / 256, 256, 0, stream>>>(w_hid, whT, DIM, HID2);
    cvt_transpose_kernel<<<(DIM * QK + 255) / 256, 256, 0, stream>>>(w_qk, wqT, DIM, QK);
    cvt_transpose_kernel<<<(HID * DIM + 255) / 256, 256, 0, stream>>>(w_out, woT, HID, DIM);

    gemm_hid_kernel<<<(ROWS / 32) * (HID2 / 64) / 8, 256, 0, stream>>>(nxb, whT, b_hid, vT, gate);
    gemm_qk_kernel<<<(ROWS / 32) * (QK / 64) / 8, 256, 0, stream>>>(nxb, wqT, b_qk, osg, osb, qbuf, kbuf);
    attn_kernel<<<BATCH * (SEQ / 16), 256, 0, stream>>>(qbuf, kbuf, vT, gate, rel, mask, omb);
    gemm_out_kernel<<<(ROWS / 32) * (DIM / 64) / 8, 256, 0, stream>>>(omb, woT, b_out, x, out);
}
